// CausalSelfAttention_43971875177006
// MI455X (gfx1250) — matrix operand order repaired and checked by reading, not compiled, not run on hardware
//
#include <hip/hip_runtime.h>

typedef __attribute__((ext_vector_type(16))) _Float16 v16h;
typedef __attribute__((ext_vector_type(8)))  _Float16 v8h;
typedef __attribute__((ext_vector_type(8)))  float    v8f;

#define S_LEN 4096
#define DM    256
#define NB    4

#define WMMA_F16(a, b, c) \
  __builtin_amdgcn_wmma_f32_16x16x32_f16(false, (a), false, (b), (short)0, (c), false, false)

// ---------- fragment loaders (ISA 7.12.2 layouts, wave32) ----------

// A-matrix 16x32 f16, row-major source, leading dim `ld` (halves).
// lane 0-15: M=lane, K = 0..7 / 16..23 ; lane 16-31: M=lane-16, K = 8..15 / 24..31
template <typename PT>
static __device__ __forceinline__ v16h load_A16(PT base, int ld, int kOff) {
  int lane = threadIdx.x & 31;
  int row  = lane & 15;
  int kb   = kOff + ((lane >> 4) << 3);
  PT p = base + row * ld + kb;
  v16h a;
  *(v8h*)&a       = *(const v8h*)p;
  *((v8h*)&a + 1) = *(const v8h*)(p + 16);
  return a;
}

// B-matrix 32x16 f16 where B[k][n] lives at base[n*ld + k].
// lane 0-15: N=lane, K = 0..7 / 16..23 ; lane 16-31: N=lane-16, K = 8..15 / 24..31
template <typename PT>
static __device__ __forceinline__ v16h load_B16(PT base, int ld, int kOff) {
  int lane = threadIdx.x & 31;
  int n    = lane & 15;
  int k0   = kOff + ((lane >> 4) << 3);
  PT p = base + n * ld + k0;
  v16h b;
  *(v8h*)&b       = *(const v8h*)p;
  *((v8h*)&b + 1) = *(const v8h*)(p + 16);
  return b;
}

// gfx1250 async global->LDS copy, 16 bytes per lane (ASYNCcnt tracked)
static __device__ __forceinline__ void async_copy_b128(unsigned lds_off, const void* gaddr) {
  asm volatile("global_load_async_to_lds_b128 %0, %1, off"
               :: "v"(lds_off), "v"((unsigned long long)gaddr)
               : "memory");
}

// ---------- kernel 1: fp32 -> f16 convert ----------
__global__ void csa_cvt_kernel(const float* __restrict__ src, _Float16* __restrict__ dst, int n) {
  int i = (blockIdx.x * blockDim.x + threadIdx.x) * 4;
  if (i + 3 < n) {
    float4 v = *(const float4*)(src + i);
    dst[i + 0] = (_Float16)v.x;
    dst[i + 1] = (_Float16)v.y;
    dst[i + 2] = (_Float16)v.z;
    dst[i + 3] = (_Float16)v.w;
  }
}

// ---------- kernel 2: QKV projection  y = x @ W^T (all f16 WMMA) ----------
// grid.x = (B*S)/16 M-tiles, grid.y = {0:Q,1:K,2:V}, block = 128 (4 waves)
__global__ __launch_bounds__(128) void csa_proj_kernel(
    const _Float16* __restrict__ xh,
    const _Float16* __restrict__ Wqh, const _Float16* __restrict__ Wkh,
    const _Float16* __restrict__ Wvh,
    _Float16* __restrict__ Qh, _Float16* __restrict__ Kh, _Float16* __restrict__ Vt) {
  int w    = threadIdx.x >> 5;
  int lane = threadIdx.x & 31;
  int n16  = lane & 15;
  int off  = (lane >> 4) << 3;
  int mbase = blockIdx.x * 16;
  int mat   = blockIdx.y;
  const _Float16* Wh = (mat == 0) ? Wqh : (mat == 1) ? Wkh : Wvh;
  const _Float16* Xb = xh + (size_t)mbase * DM;

  v16h af[8];
#pragma unroll
  for (int kk = 0; kk < 8; ++kk) af[kk] = load_A16(Xb, DM, kk * 32);

  for (int t = 0; t < 4; ++t) {
    int nbase = (w * 4 + t) * 16;
    v8f acc;
#pragma unroll
    for (int i = 0; i < 8; ++i) acc[i] = 0.0f;
#pragma unroll
    for (int kk = 0; kk < 8; ++kk) {
      v16h bf = load_B16(Wh + (size_t)nbase * DM, DM, kk * 32);
      acc = WMMA_F16(af[kk], bf, acc);
    }
    if (mat == 0) {       // Q with 1/sqrt(D) folded in
#pragma unroll
      for (int r = 0; r < 8; ++r)
        Qh[(size_t)(mbase + r + off) * DM + nbase + n16] = (_Float16)(acc[r] * 0.0625f);
    } else if (mat == 1) {
#pragma unroll
      for (int r = 0; r < 8; ++r)
        Kh[(size_t)(mbase + r + off) * DM + nbase + n16] = (_Float16)acc[r];
    } else {              // V stored transposed per batch: Vt[b][d][s]
      int bidx = mbase / S_LEN;
      int sbase = mbase - bidx * S_LEN;
      v8h v;
#pragma unroll
      for (int r = 0; r < 8; ++r) v[r] = (_Float16)acc[r];
      *(v8h*)(Vt + (size_t)bidx * DM * S_LEN + (size_t)(nbase + n16) * S_LEN + sbase + off) = v;
    }
  }
}

// ---------- kernel 3: causal flash attention ----------
// block = 128 (4 waves) owns 64 query rows; waves cooperatively stage the
// shared K/Vt tiles into LDS via async-to-LDS, then each wave does its own
// 16-row online-softmax WMMA pipeline. grid = (S/64, B).
__global__ __launch_bounds__(128) void csa_attn_kernel(
    const _Float16* __restrict__ Qh, const _Float16* __restrict__ Kh,
    const _Float16* __restrict__ Vt, float* __restrict__ out) {
  __shared__ _Float16 Kbuf[32 * DM];       // 16 KB: rows kv, cols d
  __shared__ _Float16 Vbuf[DM * 32];       // 16 KB: rows d, cols kv
  __shared__ _Float16 Pbuf[4][16 * 32];    //  4 KB

  int tid  = threadIdx.x;
  int w    = tid >> 5;
  int lane = tid & 31;
  int n16  = lane & 15;
  int off  = (lane >> 4) << 3;
  int b     = blockIdx.y;
  int qtb   = blockIdx.x * 64;
  int qbase = qtb + w * 16;

  const _Float16* Qb = Qh + (size_t)(b * S_LEN + qbase) * DM;
  const _Float16* Kb = Kh + (size_t)b * S_LEN * DM;
  const _Float16* Vb = Vt + (size_t)b * DM * S_LEN;

  unsigned kLds = (unsigned)(unsigned long long)(const void*)Kbuf;
  unsigned vLds = (unsigned)(unsigned long long)(const void*)Vbuf;

  v8f o[16];
#pragma unroll
  for (int n = 0; n < 16; ++n)
#pragma unroll
    for (int i = 0; i < 8; ++i) o[n][i] = 0.0f;

  float m[8], l[8];
#pragma unroll
  for (int r = 0; r < 8; ++r) { m[r] = -1e30f; l[r] = 0.0f; }

  _Float16* P = &Pbuf[w][0];
  int nblk = (qtb + 64 + 31) >> 5;   // uniform across the block

  for (int j = 0; j < nblk; ++j) {
    int kvb = j * 32;

    __syncthreads();  // previous iteration's LDS reads are done
    // --- cooperative async staging: K tile 32x256, Vt slice 256x32 ---
#pragma unroll
    for (int c = 0; c < 8; ++c) {
      int e = (c * 128 + tid) * 8;   // half index into the 32x256 tile
      async_copy_b128(kLds + e * 2, Kb + (size_t)kvb * DM + e);
    }
#pragma unroll
    for (int d2 = 0; d2 < 2; ++d2) {
      int d = tid * 2 + d2;
#pragma unroll
      for (int c = 0; c < 4; ++c)
        async_copy_b128(vLds + (d * 32 + c * 8) * 2, Vb + (size_t)d * S_LEN + kvb + c * 8);
    }
    asm volatile("s_wait_asynccnt 0" ::: "memory");
    __syncthreads();  // tiles visible to all waves

    if (kvb <= qbase + 15) {   // causal: this wave has live columns here
      v8f sA, sB;
#pragma unroll
      for (int i = 0; i < 8; ++i) { sA[i] = 0.0f; sB[i] = 0.0f; }

      // S = Q * K^T over d=256 (Q pre-scaled by 1/sqrt(D))
#pragma unroll
      for (int kk = 0; kk < 8; ++kk) {
        v16h qa = load_A16(Qb, DM, kk * 32);
        v16h bA = load_B16((const _Float16*)Kbuf, DM, kk * 32);
        v16h bB = load_B16((const _Float16*)Kbuf + 16 * DM, DM, kk * 32);
        sA = WMMA_F16(qa, bA, sA);
        sB = WMMA_F16(qa, bB, sB);
      }

      // causal mask + online softmax (row reductions over 16-lane groups)
#pragma unroll
      for (int r = 0; r < 8; ++r) {
        int row = qbase + r + off;
        float a  = sA[r];
        float bb = sB[r];
        if (kvb + n16 > row)      a  = -1e30f;
        if (kvb + 16 + n16 > row) bb = -1e30f;
        float mx = fmaxf(a, bb);
        mx = fmaxf(mx, __shfl_xor(mx, 1, 32));
        mx = fmaxf(mx, __shfl_xor(mx, 2, 32));
        mx = fmaxf(mx, __shfl_xor(mx, 4, 32));
        mx = fmaxf(mx, __shfl_xor(mx, 8, 32));
        float mnew = fmaxf(m[r], mx);
        float corr = __expf(m[r] - mnew);
        float pa = __expf(a - mnew);
        float pb = __expf(bb - mnew);
        float sum = pa + pb;
        sum += __shfl_xor(sum, 1, 32);
        sum += __shfl_xor(sum, 2, 32);
        sum += __shfl_xor(sum, 4, 32);
        sum += __shfl_xor(sum, 8, 32);
        l[r] = l[r] * corr + sum;
        m[r] = mnew;
#pragma unroll
        for (int n = 0; n < 16; ++n) o[n][r] *= corr;
        P[(r + off) * 32 + n16]      = (_Float16)pa;
        P[(r + off) * 32 + 16 + n16] = (_Float16)pb;
      }

      // wave-synchronous LDS: P stores -> A-fragment reads
      asm volatile("s_wait_dscnt 0" ::: "memory");

      v16h ap = load_A16((const _Float16*)P, 32, 0);
#pragma unroll
      for (int n = 0; n < 16; ++n) {
        v16h bv = load_B16((const _Float16*)Vbuf + (n * 16) * 32, 32, 0);
        o[n] = WMMA_F16(ap, bv, o[n]);
      }
      asm volatile("s_wait_dscnt 0" ::: "memory");
    }
  }

  float* Ob = out + (size_t)(b * S_LEN + qbase) * DM;
#pragma unroll
  for (int r = 0; r < 8; ++r) {
    float invl = 1.0f / l[r];
#pragma unroll
    for (int n = 0; n < 16; ++n)
      Ob[(size_t)(r + off) * DM + n * 16 + n16] = o[n][r] * invl;
  }
}

// ---------- host launcher ----------
extern "C" void kernel_launch(void* const* d_in, const int* in_sizes, int n_in,
                              void* d_out, int out_size, void* d_ws, size_t ws_size,
                              hipStream_t stream) {
  const float* x  = (const float*)d_in[0];
  const float* Wq = (const float*)d_in[1];
  const float* Wk = (const float*)d_in[2];
  const float* Wv = (const float*)d_in[3];
  float* out = (float*)d_out;

  const size_t nx = (size_t)NB * S_LEN * DM;  // 4,194,304
  const size_t nw = (size_t)DM * DM;          // 65,536
  _Float16* xh  = (_Float16*)d_ws;            // 8 MB
  _Float16* Qh  = xh + nx;                    // 8 MB (pre-scaled by 1/sqrt(D))
  _Float16* Kh  = Qh + nx;                    // 8 MB
  _Float16* Vt  = Kh + nx;                    // 8 MB, [b][d][s]
  _Float16* Wqh = Vt + nx;                    // 128 KB each
  _Float16* Wkh = Wqh + nw;
  _Float16* Wvh = Wkh + nw;

  // 1) f32 -> f16 conversions
  csa_cvt_kernel<<<(int)(nx / 1024), 256, 0, stream>>>(x, xh, (int)nx);
  csa_cvt_kernel<<<(int)(nw / 1024), 256, 0, stream>>>(Wq, Wqh, (int)nw);
  csa_cvt_kernel<<<(int)(nw / 1024), 256, 0, stream>>>(Wk, Wkh, (int)nw);
  csa_cvt_kernel<<<(int)(nw / 1024), 256, 0, stream>>>(Wv, Wvh, (int)nw);

  // 2) QKV projections via WMMA
  dim3 pg(NB * S_LEN / 16, 3);
  csa_proj_kernel<<<pg, 128, 0, stream>>>(xh, Wqh, Wkh, Wvh, Qh, Kh, Vt);

  // 3) causal flash attention via WMMA + async-to-LDS staging
  dim3 ag(S_LEN / 64, NB);
  csa_attn_kernel<<<ag, 128, 0, stream>>>(Qh, Kh, Vt, out);
}
